// Vector_Quantization_55542517071905
// MI455X (gfx1250) — matrix operand order repaired and checked by reading, not compiled, not run on hardware
//
#include <hip/hip_runtime.h>

// Problem constants
#define BB 16
#define CC 256
#define HH 32
#define WW 32
#define NN (BB*HH*WW)        // 16384 z rows
#define KK 8192              // codebook entries
#define NT_K (KK/16)         // 512 code tiles
#define SSTEPS (CC/32)       // 8 k-steps of 32 along C

typedef __bf16 bf16;
typedef __attribute__((ext_vector_type(16))) __bf16 v16bf;
typedef __attribute__((ext_vector_type(8)))  float  v8f;
typedef __attribute__((ext_vector_type(4)))  float  v4f;

// ---------------------------------------------------------------------------
// Prep z: transpose [B,C,H,W]->[N,C] (output z, non-temporal), and emit bf16
// hi/lo splits in WMMA B-matrix fragment order:
//   slot = ((zt*8+s)*32 + lane)*16 + e
// B 32x16 bf16 layout: lanes 0-15 hold K=0..7,16..23; lanes 16-31 K=8..15,24..31.
// ---------------------------------------------------------------------------
__global__ void prep_z_kernel(const float* __restrict__ ze, float* __restrict__ z_out,
                              bf16* __restrict__ zf_hi, bf16* __restrict__ zf_lo) {
  int idx = blockIdx.x * 256 + threadIdx.x;    // linear index into z_e [B,C,H,W]
  int hw = idx & 1023;
  int c  = (idx >> 10) & 255;
  int b  = idx >> 18;
  float x = ze[idx];
  int n = (b << 10) | hw;                      // row in [N]
  __builtin_nontemporal_store(x, z_out + (long)n * CC + c);
  bf16 hi = (bf16)x;
  bf16 lo = (bf16)(x - (float)hi);
  int zt = n >> 4, m = n & 15;
  int s  = c >> 5, k = c & 31;
  int half = (k >> 3) & 1, e = (k & 7) | ((k >> 4) << 3);
  long slot = (((long)zt * SSTEPS + s) * 32 + (m + (half << 4))) * 16 + e;
  zf_hi[slot] = hi; zf_lo[slot] = lo;
}

// ---------------------------------------------------------------------------
// Prep embedding: bf16 hi/lo splits in WMMA A-matrix fragment order.
// A 16x32 bf16 layout: lane l row M=l%16; l<16: K=0..7 -> e=0..7, K=16..23 -> e=8..15
//                                        l>=16: K=8..15 -> e=0..7, K=24..31 -> e=8..15
// ---------------------------------------------------------------------------
__global__ void prep_e_kernel(const float* __restrict__ emb,
                              bf16* __restrict__ ef_hi, bf16* __restrict__ ef_lo) {
  int idx = blockIdx.x * 256 + threadIdx.x;    // code*CC + c
  int code = idx >> 8, c = idx & 255;
  float x = emb[idx];
  bf16 hi = (bf16)x;
  bf16 lo = (bf16)(x - (float)hi);
  int t = code >> 4, m = code & 15;
  int s = c >> 5,  k = c & 31;
  int half, e;
  if      (k <  8) { half = 0; e = k;      }
  else if (k < 16) { half = 1; e = k - 8;  }
  else if (k < 24) { half = 0; e = k - 8;  }
  else             { half = 1; e = k - 16; }
  long slot = (((long)t * SSTEPS + s) * 32 + (m + (half << 4))) * 16 + e;
  ef_hi[slot] = hi; ef_lo[slot] = lo;
}

// ||e_k||^2 per code (exact f32)
__global__ void emb_norm_kernel(const float* __restrict__ emb, float* __restrict__ esq) {
  int k = blockIdx.x * 256 + threadIdx.x;
  const v4f* row = (const v4f*)(emb + (long)k * CC);
  float s = 0.f;
#pragma unroll 8
  for (int i = 0; i < CC / 4; i++) {
    v4f v = row[i];
    s += v[0] * v[0] + v[1] * v[1] + v[2] * v[2] + v[3] * v[3];
  }
  esq[k] = s;
}

// ---------------------------------------------------------------------------
// Async 8KB tile copy (global -> LDS) via CDNA5 async-LDS path (ASYNCcnt).
// 512 x 16B transfers, 128 threads, 4 chunks each; GVS addressing form.
// ---------------------------------------------------------------------------
__device__ __forceinline__ void async_copy_8k(unsigned lds_base, const bf16* __restrict__ gbase,
                                              long tile, int tid) {
  unsigned g0 = (unsigned)((tile * 512 + tid) * 16);
#pragma unroll
  for (int i = 0; i < 4; i++) {
    unsigned laddr = lds_base + (unsigned)((tid + i * 128) * 16);
    unsigned gaddr = g0 + (unsigned)(i * 128 * 16);
    asm volatile("global_load_async_to_lds_b128 %0, %1, %2"
                 :: "v"(laddr), "v"(gaddr), "s"(gbase)
                 : "memory");
  }
}

__device__ __forceinline__ void wait_async0() {
  asm volatile("s_wait_asynccnt 0x0" ::: "memory");
}

// ---------------------------------------------------------------------------
// Main: per wave, one 16-row z-tile (B fragments resident in VGPRs), stream all
// 512 code tiles through async double-buffered LDS. bf16x3 split GEMM, f32
// accum, per-lane running argmin of (||e||^2 - 2 z.e).
// ---------------------------------------------------------------------------
__launch_bounds__(128)
__global__ void vq_main_kernel(const bf16* __restrict__ zf_hi, const bf16* __restrict__ zf_lo,
                               const bf16* __restrict__ ef_hi, const bf16* __restrict__ ef_lo,
                               const float* __restrict__ esq, const float* __restrict__ emb,
                               float* __restrict__ zq_out, float* __restrict__ idx_out,
                               int* __restrict__ idx_ws) {
  __shared__ uint4 s_eh[2][512];   // 8 KB per buffer (hi)
  __shared__ uint4 s_el[2][512];   // 8 KB per buffer (lo)
  __shared__ float s_q[KK];        // 32 KB ||e||^2

  int tid  = threadIdx.x;
  int lane = tid & 31;
  int wave = tid >> 5;
  int zt   = blockIdx.x * 4 + wave;

  unsigned eh_lds = (unsigned)(uintptr_t)&s_eh[0][0];
  unsigned el_lds = (unsigned)(uintptr_t)&s_el[0][0];

  // async preload code tile 0 into buffer 0
  async_copy_8k(eh_lds, ef_hi, 0, tid);
  async_copy_8k(el_lds, ef_lo, 0, tid);

  // esq -> LDS (plain loads, overlap with async)
  {
    const v4f* g = (const v4f*)esq;
    v4f* sh = (v4f*)s_q;
    for (int i = tid; i < KK / 4; i += 128) sh[i] = g[i];
  }

  // resident B fragments (z tile): 8 k-steps x {hi,lo}
  v16bf bh[8], bl[8];
  {
    const v16bf* zh = (const v16bf*)zf_hi;
    const v16bf* zl = (const v16bf*)zf_lo;
#pragma unroll
    for (int s = 0; s < 8; s++) {
      long o = ((long)zt * 8 + s) * 32 + lane;
      bh[s] = zh[o];
      bl[s] = zl[o];
    }
  }
  wait_async0();
  __syncthreads();

  float minv[8]; int mini[8];
#pragma unroll
  for (int r = 0; r < 8; r++) { minv[r] = 3.4e38f; mini[r] = 0; }

  int cur = 0;
  for (int t = 0; t < NT_K; t++) {
    int nxt = cur ^ 1;
    // kick off async fetch of tile t+1 into the other buffer (safe: all waves
    // finished reading it before the barrier that ended iteration t-1)
    if (t + 1 < NT_K) {
      async_copy_8k(eh_lds + (unsigned)nxt * 8192u, ef_hi, t + 1, tid);
      async_copy_8k(el_lds + (unsigned)nxt * 8192u, ef_lo, t + 1, tid);
    }

    const bf16* eh = (const bf16*)s_eh[cur];
    const bf16* el = (const bf16*)s_el[cur];
    v8f acc0 = {}; v8f acc1 = {};
#pragma unroll
    for (int s = 0; s < 8; s += 2) {
      v16bf ah0 = *(const v16bf*)(eh + ((s    ) * 32 + lane) * 16);
      v16bf al0 = *(const v16bf*)(el + ((s    ) * 32 + lane) * 16);
      v16bf ah1 = *(const v16bf*)(eh + ((s + 1) * 32 + lane) * 16);
      v16bf al1 = *(const v16bf*)(el + ((s + 1) * 32 + lane) * 16);
      // bf16x3: hi*hi + hi*lo + lo*hi, two independent accumulator chains
      acc0 = __builtin_amdgcn_wmma_f32_16x16x32_bf16(false, ah0, false, bh[s    ], (short)0, acc0, false, false);
      acc1 = __builtin_amdgcn_wmma_f32_16x16x32_bf16(false, ah1, false, bh[s + 1], (short)0, acc1, false, false);
      acc0 = __builtin_amdgcn_wmma_f32_16x16x32_bf16(false, ah0, false, bl[s    ], (short)0, acc0, false, false);
      acc1 = __builtin_amdgcn_wmma_f32_16x16x32_bf16(false, ah1, false, bl[s + 1], (short)0, acc1, false, false);
      acc0 = __builtin_amdgcn_wmma_f32_16x16x32_bf16(false, al0, false, bh[s    ], (short)0, acc0, false, false);
      acc1 = __builtin_amdgcn_wmma_f32_16x16x32_bf16(false, al1, false, bh[s + 1], (short)0, acc1, false, false);
    }

    // running argmin: this lane's 8 codes for this tile
    int cbase = (t << 4) + ((lane >> 4) << 3);
    v4f q0 = *(const v4f*)(s_q + cbase);
    v4f q1 = *(const v4f*)(s_q + cbase + 4);
    float d;
    d = q0[0] - 2.0f * (acc0[0] + acc1[0]); if (d < minv[0]) { minv[0] = d; mini[0] = cbase + 0; }
    d = q0[1] - 2.0f * (acc0[1] + acc1[1]); if (d < minv[1]) { minv[1] = d; mini[1] = cbase + 1; }
    d = q0[2] - 2.0f * (acc0[2] + acc1[2]); if (d < minv[2]) { minv[2] = d; mini[2] = cbase + 2; }
    d = q0[3] - 2.0f * (acc0[3] + acc1[3]); if (d < minv[3]) { minv[3] = d; mini[3] = cbase + 3; }
    d = q1[0] - 2.0f * (acc0[4] + acc1[4]); if (d < minv[4]) { minv[4] = d; mini[4] = cbase + 4; }
    d = q1[1] - 2.0f * (acc0[5] + acc1[5]); if (d < minv[5]) { minv[5] = d; mini[5] = cbase + 5; }
    d = q1[2] - 2.0f * (acc0[6] + acc1[6]); if (d < minv[6]) { minv[6] = d; mini[6] = cbase + 6; }
    d = q1[3] - 2.0f * (acc0[7] + acc1[7]); if (d < minv[7]) { minv[7] = d; mini[7] = cbase + 7; }

    // drain this wave's async writes, then make them visible to all waves
    wait_async0();
    __syncthreads();
    cur = nxt;
  }

  // reduce the 8 per-lane candidates (codes ascending -> strict < keeps lowest)
  float bv = minv[0]; int bi = mini[0];
#pragma unroll
  for (int r = 1; r < 8; r++) {
    if (minv[r] < bv || (minv[r] == bv && mini[r] < bi)) { bv = minv[r]; bi = mini[r]; }
  }
  // merge lane halves (lane l <-> l+16 hold codes %16 in [0,8) vs [8,16))
  float ov = __shfl_xor(bv, 16, 32);
  int   oi = __shfl_xor(bi, 16, 32);
  if (ov < bv || (ov == bv && oi < bi)) { bv = ov; bi = oi; }

  if (lane < 16) {
    int n = zt * 16 + lane;
    idx_ws[n]  = bi;
    idx_out[n] = (float)bi;
  }
  // cooperative z_q gather: 16 rows x 256 floats per wave (NT stores)
  for (int r = 0; r < 16; r++) {
    int bir = __shfl(bi, r, 32);
    const v4f* src = (const v4f*)(emb + (long)bir * CC);
    v4f* dst = (v4f*)(zq_out + ((long)zt * 16 + r) * CC);
    __builtin_nontemporal_store(src[lane],      dst + lane);
    __builtin_nontemporal_store(src[lane + 32], dst + lane + 32);
  }
}

// one-hot: 536 MB of v4f non-temporal streaming stores (the HBM-bound tail);
// NT keeps the 192MB L2 free for the embedding working set.
__global__ void onehot_kernel(const int* __restrict__ idx_ws, v4f* __restrict__ oh) {
  int i = blockIdx.x * 256 + threadIdx.x;   // one v4f per thread, N*K/4 total
  int row = i >> 11;                        // K/4 = 2048 v4f per row
  int idx = idx_ws[row];
  int c = (i & 2047) << 2;
  v4f v = {0.f, 0.f, 0.f, 0.f};
  int d = idx - c;
  if ((unsigned)d < 4u) v[d] = 1.0f;
  __builtin_nontemporal_store(v, oh + i);
}

extern "C" void kernel_launch(void* const* d_in, const int* in_sizes, int n_in,
                              void* d_out, int out_size, void* d_ws, size_t ws_size,
                              hipStream_t stream) {
  const float* ze  = (const float*)d_in[0];   // [B,C,H,W]
  const float* emb = (const float*)d_in[1];   // [K,C]

  float* out     = (float*)d_out;
  float* z_out   = out;                          // N*C
  float* zq_out  = out + (long)NN * CC;          // N*C
  float* idx_out = out + (long)2 * NN * CC;      // N (as float)
  float* oh_out  = out + (long)2 * NN * CC + NN; // N*K

  char* ws = (char*)d_ws;
  bf16*  zf_hi = (bf16*)(ws);                    //  8 MB
  bf16*  zf_lo = (bf16*)(ws + 8388608);          //  8 MB
  bf16*  ef_hi = (bf16*)(ws + 16777216);         //  4 MB
  bf16*  ef_lo = (bf16*)(ws + 20971520);         //  4 MB
  float* esq   = (float*)(ws + 25165824);        // 32 KB
  int*   idxws = (int*)  (ws + 25198592);        // 64 KB

  prep_e_kernel  <<<KK * CC / 256, 256, 0, stream>>>(emb, ef_hi, ef_lo);
  emb_norm_kernel<<<KK / 256,      256, 0, stream>>>(emb, esq);
  prep_z_kernel  <<<NN * CC / 256, 256, 0, stream>>>(ze, z_out, zf_hi, zf_lo);
  vq_main_kernel <<<NN / 64,       128, 0, stream>>>(zf_hi, zf_lo, ef_hi, ef_lo,
                                                     esq, emb, zq_out, idx_out, idxws);
  onehot_kernel  <<<(long)NN * KK / 4 / 256, 256, 0, stream>>>(idxws, (v4f*)oh_out);
}
